// SuperPointJAX_7653631721976
// MI455X (gfx1250) — matrix operand order repaired and checked by reading, not compiled, not run on hardware
//
#include <hip/hip_runtime.h>
#include <hip/hip_bf16.h>
#include <cstdint>
#include <cstddef>

typedef _Float16 half_t;
typedef __attribute__((ext_vector_type(16))) _Float16 v16h;
typedef __attribute__((ext_vector_type(8)))  float    v8f;

#define BN_SCALE 0.9995003746879392f
#define THRESH   0.005f
#define CAP      16384
#define TOPK     1024

// ---------------------------------------------------------------------------
// fp32 -> fp16 weight packing
// ---------------------------------------------------------------------------
__global__ void k_f32_to_f16(const float* __restrict__ src, half_t* __restrict__ dst, int n) {
    int i = blockIdx.x * blockDim.x + threadIdx.x;
    if (i < n) dst[i] = (half_t)src[i];
}

// ---------------------------------------------------------------------------
// Layer 1: 3x3 conv, Cin=1 -> Cout=64 (K=9 is too small for WMMA; direct VALU)
// ---------------------------------------------------------------------------
__global__ void conv1_direct(const float* __restrict__ img, const half_t* __restrict__ w,
                             const float* __restrict__ bias, half_t* __restrict__ out,
                             int H, int W) {
    const int HW = H * W;
    int i = blockIdx.x * blockDim.x + threadIdx.x;
    int total = 4 * HW * 64;
    if (i >= total) return;
    int co  = i & 63;
    int pix = (i >> 6) % HW;
    int bz  = i / (HW * 64);
    int y = pix / W, x = pix % W;
    float acc = bias[co];
    #pragma unroll
    for (int ky = 0; ky < 3; ++ky) {
        #pragma unroll
        for (int kx = 0; kx < 3; ++kx) {
            int yy = y + ky - 1, xx = x + kx - 1;
            if (yy >= 0 && yy < H && xx >= 0 && xx < W)
                acc += img[(size_t)bz * HW + yy * W + xx] * (float)w[(ky * 3 + kx) * 64 + co];
        }
    }
    acc = fmaxf(acc, 0.f) * BN_SCALE;
    out[i] = (half_t)acc;
}

// ---------------------------------------------------------------------------
// Implicit-GEMM conv via V_WMMA_F32_16X16X32_F16 (wave32, CDNA5).
// Block = 256 thr = 8 waves; tile = 64 output pixels x 64 output channels.
// Wave (mw, nw): M16 subtile at mw*16, N32 at nw*32 -> two 16x16 f32 accs.
// A tile staged global->LDS with GLOBAL_LOAD_ASYNC_TO_LDS_B128 (ASYNCcnt);
// zero-pad lanes fall back to ds_store_b128 under divergence (async loads
// are EXEC-gated). B tile staged transposed via VGPRs for contiguous
// fragment gathers. Requires H*W%64==0, Cin%32==0, Cout%64==0, ks in {1,3}.
// Weights packed HWIO fp16: w[(tap*Cin + ci)*Cout + co].
// ---------------------------------------------------------------------------
__global__ __launch_bounds__(256, 2)
void conv_wmma(const half_t* __restrict__ in, const half_t* __restrict__ w,
               const float* __restrict__ bias, half_t* __restrict__ out,
               int H, int W, int Cin, int Cout, int ks, int relu) {
    __shared__ half_t As[64][40];   // [pixel][ci]   (row stride 80B, 16B aligned)
    __shared__ half_t Bs[64][40];   // [co][ci] transposed weight tile

    const int bz       = blockIdx.z;
    const int tileBase = blockIdx.x * 64;
    const int coBase   = blockIdx.y * 64;
    const int tid  = threadIdx.x;
    const int lane = tid & 31;
    const int wid  = tid >> 5;
    const int m0 = (wid & 3) * 16;
    const int n0 = (wid >> 2) * 32;
    const int l  = lane & 15;
    const int hi = lane >> 4;
    const int HW = H * W;
    const half_t* inB = in + (size_t)bz * HW * Cin;

    v8f acc0 = {};
    v8f acc1 = {};

    const int taps = ks * ks;
    const int rad  = ks >> 1;

    // A-staging assignment: thread -> (pixel, ci-chunk of 8 halves = 16B)
    const int pA = tid >> 2;
    const int cA = (tid & 3) * 8;
    const int py = (tileBase + pA) / W;
    const int px = (tileBase + pA) % W;
    const unsigned ldsA = (unsigned)(uintptr_t)&As[pA][cA];
    // B-staging assignment: thread -> (ci row, co-chunk of 8 halves)
    const int rB = tid >> 3;
    const int cB = (tid & 7) * 8;

    for (int t = 0; t < taps; ++t) {
        const int dy = t / ks - rad;
        const int dx = t % ks - rad;
        const int yy = py + dy, xx = px + dx;
        const bool inb = (yy >= 0 && yy < H && xx >= 0 && xx < W);
        const half_t* srcW = w + (size_t)t * Cin * Cout;

        for (int kb = 0; kb < Cin; kb += 32) {
            // ---- stage A tile (64 x 32 halves) via async global->LDS DMA
            if (inb) {
                unsigned aoff = (unsigned)((((size_t)yy * W + xx) * Cin + kb + cA) * 2);
                asm volatile("global_load_async_to_lds_b128 %0, %1, %2"
                             :: "v"(ldsA), "v"(aoff), "s"(inB)
                             : "memory");
            } else {
                *(int4*)&As[pA][cA] = make_int4(0, 0, 0, 0);   // zero-pad halo
            }
            // ---- stage B tile transposed: Bs[co][ci]
            {
                const int4 v = *(const int4*)(srcW + (size_t)(kb + rB) * Cout + coBase + cB);
                const half_t* hv = (const half_t*)&v;
                #pragma unroll
                for (int j = 0; j < 8; ++j) Bs[cB + j][rB] = hv[j];
            }
            if (kb + 32 < Cin)  // warm GL2 for next K-block of weights
                __builtin_prefetch((const void*)(srcW + (size_t)(kb + 32 + rB) * Cout + coBase + cB), 0, 0);
            asm volatile("s_wait_asynccnt 0" ::: "memory");    // LDS writes from DMA visible
            __syncthreads();

            // ---- load fragments; A and B share contiguous k order (k = 16*hi + s)
            v16h a, b0, b1;
            #pragma unroll
            for (int j = 0; j < 8; ++j) {            // A 16x32: lane<16 K0..15, lane>=16 K16..31
                a[j]     = As[m0 + l][hi * 16 + j];
                a[8 + j] = As[m0 + l][hi * 16 + 8 + j];
            }
            #pragma unroll
            for (int j = 0; j < 16; ++j) {           // B 32x16: N=lane%16, lane<16 K0..15, lane>=16 K16..31
                b0[j] = Bs[n0 + l][hi * 16 + j];
                b1[j] = Bs[n0 + 16 + l][hi * 16 + j];
            }
            acc0 = __builtin_amdgcn_wmma_f32_16x16x32_f16(false, a, false, b0, (short)0, acc0, false, false);
            acc1 = __builtin_amdgcn_wmma_f32_16x16x32_f16(false, a, false, b1, (short)0, acc1, false, false);
            __syncthreads();
        }
    }

    // ---- epilogue: bias + optional ReLU + BN scale, fp16 NHWC store
    const float bi0 = bias[coBase + n0 + l];
    const float bi1 = bias[coBase + n0 + 16 + l];
    half_t* outB = out + (size_t)bz * HW * Cout;
    #pragma unroll
    for (int r = 0; r < 8; ++r) {
        int m   = m0 + hi * 8 + r;   // C/D layout: lanes<16 M=r, lanes>=16 M=8+r
        int pix = tileBase + m;
        float v0 = acc0[r] + bi0;
        float v1 = acc1[r] + bi1;
        if (relu) { v0 = fmaxf(v0, 0.f); v1 = fmaxf(v1, 0.f); }
        v0 *= BN_SCALE; v1 *= BN_SCALE;
        outB[(size_t)pix * Cout + coBase + n0 + l]      = (half_t)v0;
        outB[(size_t)pix * Cout + coBase + n0 + 16 + l] = (half_t)v1;
    }
}

// ---------------------------------------------------------------------------
// 2x2 maxpool, NHWC fp16
// ---------------------------------------------------------------------------
__global__ void maxpool2(const half_t* __restrict__ in, half_t* __restrict__ out,
                         int Hi, int Wi, int Ho, int Wo, int C) {
    int i = blockIdx.x * blockDim.x + threadIdx.x;
    int total = 4 * Ho * Wo * C;
    if (i >= total) return;
    int c = i % C; int r = i / C;
    int xo = r % Wo; r /= Wo;
    int yo = r % Ho; int bz = r / Ho;
    const half_t* p = in + (((size_t)bz * Hi + 2 * yo) * Wi + 2 * xo) * C + c;
    float m = fmaxf(fmaxf((float)p[0], (float)p[C]),
                    fmaxf((float)p[(size_t)Wi * C], (float)p[(size_t)Wi * C + C]));
    out[i] = (half_t)m;
}

// ---------------------------------------------------------------------------
// Detector 1x1 conv 256 -> 65 (Cout=65 not WMMA-tileable; tiny: 80 MFLOP)
// ---------------------------------------------------------------------------
__global__ void conv1x1_65(const half_t* __restrict__ in, const half_t* __restrict__ w,
                           const float* __restrict__ bias, float* __restrict__ logits) {
    int i = blockIdx.x * blockDim.x + threadIdx.x;
    int total = 4 * 4800 * 65;
    if (i >= total) return;
    int co = i % 65; int p = i / 65;
    const half_t* ip = in + (size_t)p * 256;
    float acc = bias[co];
    for (int ci = 0; ci < 256; ++ci) acc += (float)ip[ci] * (float)w[ci * 65 + co];
    logits[i] = acc * BN_SCALE;   // relu=False in reference
}

// ---------------------------------------------------------------------------
// Softmax over 65 + drop dustbin + pixel-shuffle (8x8) to full-res score map
// ---------------------------------------------------------------------------
__global__ void softmax_shuffle(const float* __restrict__ logits, float* __restrict__ s,
                                int Hp, int Wp, int H, int W) {
    int i = blockIdx.x * blockDim.x + threadIdx.x;
    int total = 4 * Hp * Wp;
    if (i >= total) return;
    const float* lp = logits + (size_t)i * 65;
    float mx = lp[0];
    for (int c = 1; c < 65; ++c) mx = fmaxf(mx, lp[c]);
    float sum = 0.f;
    for (int c = 0; c < 65; ++c) sum += __expf(lp[c] - mx);
    float inv = 1.f / sum;
    int bz = i / (Hp * Wp); int pp = i % (Hp * Wp);
    int hp = pp / Wp, wp = pp % Wp;
    for (int c = 0; c < 64; ++c) {
        int y = hp * 8 + (c >> 3);
        int x = wp * 8 + (c & 7);
        s[(size_t)bz * H * W + y * W + x] = __expf(lp[c] - mx) * inv;
    }
}

// ---------------------------------------------------------------------------
// Separable 9-wide max pooling (radius 4, clipped window == -inf padding)
// ---------------------------------------------------------------------------
__global__ void pool9h(const float* __restrict__ src, float* __restrict__ dst, int H, int W) {
    int i = blockIdx.x * blockDim.x + threadIdx.x;
    int HW = H * W; int total = 4 * HW;
    if (i >= total) return;
    int pp = i % HW; int y = pp / W, x = pp % W;
    const float* row = src + (size_t)(i / HW) * HW + (size_t)y * W;
    float m = -3.0e38f;
    for (int dx = -4; dx <= 4; ++dx) {
        int xx = x + dx;
        if (xx >= 0 && xx < W) m = fmaxf(m, row[xx]);
    }
    dst[i] = m;
}
__global__ void pool9v(const float* __restrict__ src, float* __restrict__ dst, int H, int W) {
    int i = blockIdx.x * blockDim.x + threadIdx.x;
    int HW = H * W; int total = 4 * HW;
    if (i >= total) return;
    int pp = i % HW; int y = pp / W, x = pp % W;
    const float* img = src + (size_t)(i / HW) * HW;
    float m = -3.0e38f;
    for (int dy = -4; dy <= 4; ++dy) {
        int yy = y + dy;
        if (yy >= 0 && yy < H) m = fmaxf(m, img[(size_t)yy * W + x]);
    }
    dst[i] = m;
}

__global__ void k_eq_mask(const float* a, const float* b, float* m, int n) {
    int i = blockIdx.x * blockDim.x + threadIdx.x;
    if (i < n) m[i] = (a[i] == b[i]) ? 1.f : 0.f;
}
__global__ void k_supp(const float* s, const float* pm, float* ss, int n) {
    int i = blockIdx.x * blockDim.x + threadIdx.x;
    if (i < n) ss[i] = (pm[i] > 0.f) ? 0.f : s[i];
}
__global__ void k_merge(float* m, const float* ss, const float* ps, const float* pm, int n) {
    int i = blockIdx.x * blockDim.x + threadIdx.x;
    if (i < n) m[i] = ((m[i] != 0.f) || ((ss[i] == ps[i]) && !(pm[i] > 0.f))) ? 1.f : 0.f;
}
__global__ void k_final_border(const float* s, const float* m, float* out, int H, int W) {
    int i = blockIdx.x * blockDim.x + threadIdx.x;
    int HW = H * W; int total = 4 * HW;
    if (i >= total) return;
    int pp = i % HW; int y = pp / W, x = pp % W;
    float v;
    if (y < 4 || y >= H - 4 || x < 4 || x >= W - 4) v = -1.f;
    else v = (m[i] != 0.f) ? s[i] : 0.f;
    out[i] = v;
}

// ---------------------------------------------------------------------------
// Candidate compaction + deterministic top-K (bitonic sort in 128KB LDS,
// legal on CDNA5: 320KB LDS per WGP)
// ---------------------------------------------------------------------------
__global__ void cand_init(float* cscore, int* cidx, int* cnt, int n) {
    int i = blockIdx.x * blockDim.x + threadIdx.x;
    if (i < n) { cscore[i] = -1e30f; cidx[i] = 0; }
    if (i < 4) cnt[i] = 0;
}
__global__ void cand_collect(const float* __restrict__ s, float* cscore, int* cidx,
                             int* cnt, int HW) {
    int i = blockIdx.x * blockDim.x + threadIdx.x;
    if (i >= 4 * HW) return;
    float v = s[i];
    if (v > THRESH) {
        int bz = i / HW;
        int p = atomicAdd(cnt + bz, 1);
        if (p < CAP) { cscore[bz * CAP + p] = v; cidx[bz * CAP + p] = i % HW; }
    }
}
__global__ __launch_bounds__(1024)
void topk_sort(const float* __restrict__ cscore, const int* __restrict__ cidx,
               float* __restrict__ out_kpts, float* __restrict__ out_scores,
               float* __restrict__ out_valid, int* __restrict__ topidx, int W) {
    extern __shared__ char smem[];
    float* ss = (float*)smem;
    int*   si = (int*)(smem + (size_t)CAP * sizeof(float));
    int bz = blockIdx.x, tid = threadIdx.x;
    for (int i = tid; i < CAP; i += 1024) { ss[i] = cscore[bz * CAP + i]; si[i] = cidx[bz * CAP + i]; }
    __syncthreads();
    for (int k = 2; k <= CAP; k <<= 1) {
        for (int j = k >> 1; j > 0; j >>= 1) {
            for (int i = tid; i < CAP; i += 1024) {
                int ixj = i ^ j;
                if (ixj > i) {
                    float s1 = ss[i], s2 = ss[ixj];
                    int   i1 = si[i], i2 = si[ixj];
                    bool fb = (s1 > s2) || (s1 == s2 && i1 < i2);   // deterministic order
                    bool up = ((i & k) == 0);
                    if (up ? !fb : fb) { ss[i] = s2; ss[ixj] = s1; si[i] = i2; si[ixj] = i1; }
                }
            }
            __syncthreads();
        }
    }
    float sc = ss[tid]; int id = si[tid];
    bool valid = sc > -1e29f;
    int o = bz * TOPK + tid;
    out_kpts[o * 2 + 0] = (float)(id / W);
    out_kpts[o * 2 + 1] = (float)(id % W);
    out_scores[o] = valid ? sc : 0.f;
    out_valid[o]  = valid ? 1.f : 0.f;
    topidx[o] = id;
}

// ---------------------------------------------------------------------------
// Descriptor bilinear sampling + L2 normalization; 1 block per keypoint.
// desc_map NHWC fp16 [4,60,80,256] -> out [4,1024,256] f32
// ---------------------------------------------------------------------------
__global__ __launch_bounds__(256)
void desc_sample(const half_t* __restrict__ dm, const int* __restrict__ topidx,
                 float* __restrict__ out_desc) {
    __shared__ float red[256];
    int kp = blockIdx.x; int bz = kp >> 10; int c = threadIdx.x;
    int id = topidx[kp];
    int ty = id / 640, tx = id % 640;
    float kx = ((float)tx + 0.5f) / 640.f * 2.f - 1.f;
    float ky = ((float)ty + 0.5f) / 480.f * 2.f - 1.f;
    float gx = (kx + 1.f) * 0.5f * 79.f;
    float gy = (ky + 1.f) * 0.5f * 59.f;
    int x0 = (int)floorf(gx), y0 = (int)floorf(gy);
    float wx = gx - (float)x0, wy = gy - (float)y0;
    int x0c = min(max(x0, 0), 79), x1c = min(max(x0 + 1, 0), 79);
    int y0c = min(max(y0, 0), 59), y1c = min(max(y0 + 1, 0), 59);
    const half_t* base = dm + (size_t)bz * 60 * 80 * 256;
    float tl = (float)base[((size_t)y0c * 80 + x0c) * 256 + c];
    float tr = (float)base[((size_t)y0c * 80 + x1c) * 256 + c];
    float bl = (float)base[((size_t)y1c * 80 + x0c) * 256 + c];
    float br = (float)base[((size_t)y1c * 80 + x1c) * 256 + c];
    float v = tl * (1.f - wx) * (1.f - wy) + tr * wx * (1.f - wy)
            + bl * (1.f - wx) * wy + br * wx * wy;
    red[c] = v * v;
    __syncthreads();
    for (int off = 128; off > 0; off >>= 1) {
        if (c < off) red[c] += red[c + off];
        __syncthreads();
    }
    float norm = sqrtf(red[0] + 1e-8f);
    out_desc[(size_t)kp * 256 + c] = v / norm;
}

// ---------------------------------------------------------------------------
extern "C" void kernel_launch(void* const* d_in, const int* in_sizes, int n_in,
                              void* d_out, int out_size, void* d_ws, size_t ws_size,
                              hipStream_t stream) {
    (void)n_in; (void)out_size; (void)ws_size;
    const float* img = (const float*)d_in[0];
    // Param order: image; bb (w,b)x8; det (w,b)x2; desc (w,b)x2
    const int wi[12] = {1, 3, 5, 7, 9, 11, 13, 15, 17, 19, 21, 23};
    const float* Wf[12]; const float* Bf[12];
    for (int i = 0; i < 12; ++i) { Wf[i] = (const float*)d_in[wi[i]]; Bf[i] = (const float*)d_in[wi[i] + 1]; }

    // bump allocator over d_ws (assumed 256B-aligned)
    char* base = (char*)d_ws;
    size_t off = 0;
    auto alloc = [&](size_t bytes) -> char* {
        char* p = base + off;
        off = (off + bytes + 255) & ~(size_t)255;
        return p;
    };
    const int HW = 480 * 640;           // 307200
    half_t* X  = (half_t*)alloc((size_t)4 * HW * 64 * 2);
    half_t* Y  = (half_t*)alloc((size_t)4 * HW * 64 * 2);
    half_t* F  = (half_t*)alloc((size_t)4 * 4800 * 128 * 2);
    half_t* D1 = (half_t*)alloc((size_t)4 * 4800 * 256 * 2);
    half_t* DM = (half_t*)alloc((size_t)4 * 4800 * 256 * 2);
    float*  LG = (float*)alloc((size_t)4 * 4800 * 65 * 4);
    float*  S  = (float*)alloc((size_t)4 * HW * 4);
    float*  M  = (float*)alloc((size_t)4 * HW * 4);
    float*  SS = (float*)alloc((size_t)4 * HW * 4);
    float*  PM = (float*)alloc((size_t)4 * HW * 4);
    float*  PS = (float*)alloc((size_t)4 * HW * 4);
    float*  T  = (float*)alloc((size_t)4 * HW * 4);
    float*  SF = (float*)alloc((size_t)4 * HW * 4);
    half_t* Wh[12];
    for (int i = 0; i < 12; ++i) Wh[i] = (half_t*)alloc((size_t)in_sizes[wi[i]] * 2);
    float* CSC = (float*)alloc((size_t)4 * CAP * 4);
    int*   CID = (int*)alloc((size_t)4 * CAP * 4);
    int*   CNT = (int*)alloc(256);
    int*   TIX = (int*)alloc((size_t)4 * TOPK * 4);

    // pack weights to fp16
    for (int i = 0; i < 12; ++i) {
        int n = in_sizes[wi[i]];
        k_f32_to_f16<<<(n + 255) / 256, 256, 0, stream>>>(Wf[i], Wh[i], n);
    }

    // ---- backbone
    { int tot = 4 * HW * 64;
      conv1_direct<<<(tot + 255) / 256, 256, 0, stream>>>(img, Wh[0], Bf[0], X, 480, 640); }
    conv_wmma<<<dim3(HW / 64, 1, 4), 256, 0, stream>>>(X, Wh[1], Bf[1], Y, 480, 640, 64, 64, 3, 1);
    { int tot = 4 * 240 * 320 * 64;
      maxpool2<<<(tot + 255) / 256, 256, 0, stream>>>(Y, X, 480, 640, 240, 320, 64); }
    conv_wmma<<<dim3(76800 / 64, 1, 4), 256, 0, stream>>>(X, Wh[2], Bf[2], Y, 240, 320, 64, 64, 3, 1);
    conv_wmma<<<dim3(76800 / 64, 1, 4), 256, 0, stream>>>(Y, Wh[3], Bf[3], X, 240, 320, 64, 64, 3, 1);
    { int tot = 4 * 120 * 160 * 64;
      maxpool2<<<(tot + 255) / 256, 256, 0, stream>>>(X, Y, 240, 320, 120, 160, 64); }
    conv_wmma<<<dim3(19200 / 64, 2, 4), 256, 0, stream>>>(Y, Wh[4], Bf[4], X, 120, 160, 64, 128, 3, 1);
    conv_wmma<<<dim3(19200 / 64, 2, 4), 256, 0, stream>>>(X, Wh[5], Bf[5], Y, 120, 160, 128, 128, 3, 1);
    { int tot = 4 * 60 * 80 * 128;
      maxpool2<<<(tot + 255) / 256, 256, 0, stream>>>(Y, X, 120, 160, 60, 80, 128); }
    conv_wmma<<<dim3(75, 2, 4), 256, 0, stream>>>(X, Wh[6], Bf[6], Y, 60, 80, 128, 128, 3, 1);
    conv_wmma<<<dim3(75, 2, 4), 256, 0, stream>>>(Y, Wh[7], Bf[7], F, 60, 80, 128, 128, 3, 1);

    // ---- detector head
    conv_wmma<<<dim3(75, 4, 4), 256, 0, stream>>>(F, Wh[8], Bf[8], D1, 60, 80, 128, 256, 3, 1);
    { int tot = 4 * 4800 * 65;
      conv1x1_65<<<(tot + 255) / 256, 256, 0, stream>>>(D1, Wh[9], Bf[9], LG); }
    { int tot = 4 * 4800;
      softmax_shuffle<<<(tot + 255) / 256, 256, 0, stream>>>(LG, S, 60, 80, 480, 640); }

    // ---- NMS (radius 4, 2 refinement iterations)
    int n = 4 * HW; int blk = (n + 255) / 256;
    pool9h<<<blk, 256, 0, stream>>>(S, T, 480, 640);
    pool9v<<<blk, 256, 0, stream>>>(T, PM, 480, 640);
    k_eq_mask<<<blk, 256, 0, stream>>>(S, PM, M, n);
    for (int it = 0; it < 2; ++it) {
        pool9h<<<blk, 256, 0, stream>>>(M, T, 480, 640);
        pool9v<<<blk, 256, 0, stream>>>(T, PM, 480, 640);
        k_supp<<<blk, 256, 0, stream>>>(S, PM, SS, n);
        pool9h<<<blk, 256, 0, stream>>>(SS, T, 480, 640);
        pool9v<<<blk, 256, 0, stream>>>(T, PS, 480, 640);
        k_merge<<<blk, 256, 0, stream>>>(M, SS, PS, PM, n);
    }
    k_final_border<<<blk, 256, 0, stream>>>(S, M, SF, 480, 640);

    // ---- deterministic top-K
    { int tot = 4 * CAP;
      cand_init<<<(tot + 255) / 256, 256, 0, stream>>>(CSC, CID, CNT, tot); }
    cand_collect<<<blk, 256, 0, stream>>>(SF, CSC, CID, CNT, HW);
    float* out = (float*)d_out;
    float* o_kpts   = out;                       // 4*1024*2
    float* o_scores = out + 8192;                // 4*1024
    float* o_desc   = out + 12288;               // 4*1024*256
    float* o_valid  = out + 12288 + 4 * TOPK * 256;
    topk_sort<<<4, 1024, (size_t)CAP * 8, stream>>>(CSC, CID, o_kpts, o_scores, o_valid, TIX, 640);

    // ---- descriptor head + sampling
    conv_wmma<<<dim3(75, 4, 4), 256, 0, stream>>>(F, Wh[10], Bf[10], D1, 60, 80, 128, 256, 3, 1);
    conv_wmma<<<dim3(75, 4, 4), 256, 0, stream>>>(D1, Wh[11], Bf[11], DM, 60, 80, 256, 256, 1, 0);
    desc_sample<<<4 * TOPK, 256, 0, stream>>>(DM, TIX, o_desc);
}
